// FeaturePropagation_22978075033667
// MI455X (gfx1250) — matrix operand order repaired and checked by reading, not compiled, not run on hardware
//
#include <hip/hip_runtime.h>
#include <hip/hip_bf16.h>

// ---------------- problem constants ----------------
#define BB   8
#define NN   8192
#define SS   2048
#define D1   256
#define D2   512
#define CIN  768          // D1 + D2
#define HH   512
#define PP   (BB * NN)    // 65536 points total
#define DIST_EPS 1e-08f
#define BN_EPS   1e-05f

typedef __attribute__((ext_vector_type(16))) __bf16 v16bf;
typedef __attribute__((ext_vector_type(8)))  __bf16 v8bf;
typedef __attribute__((ext_vector_type(8)))  float  v8f;
typedef __attribute__((ext_vector_type(4)))  unsigned int u32x4;
typedef __attribute__((ext_vector_type(8)))  int i32x8;
typedef __attribute__((ext_vector_type(4)))  int i32x4;

#if __has_builtin(__builtin_amdgcn_tensor_load_to_lds) && \
    __has_builtin(__builtin_amdgcn_s_wait_tensorcnt)
#define USE_TDM 1
#else
#define USE_TDM 0
#endif

// fp32 -> bf16 (round to nearest even), stored as raw ushort bits
__device__ __forceinline__ unsigned short f2bf(float f) {
    unsigned int u = __float_as_uint(f);
    u += 0x7FFFu + ((u >> 16) & 1u);
    return (unsigned short)(u >> 16);
}

// ---------------- weight fp32 -> bf16 ----------------
__global__ void cvt_bf16_kernel(const float* __restrict__ src,
                                unsigned short* __restrict__ dst, int n) {
    for (int i = blockIdx.x * blockDim.x + threadIdx.x; i < n;
         i += gridDim.x * blockDim.x)
        dst[i] = f2bf(src[i]);
}

// ---------------- fused 3-NN interpolation + concat ----------------
__global__ __launch_bounds__(256)
void knn_interp_kernel(const float* __restrict__ xyz1,   // (B,3,N)
                       const float* __restrict__ xyz2,   // (B,3,S)
                       const float* __restrict__ points1,// (B,D1,N)
                       const float* __restrict__ points2,// (B,D2,S)
                       unsigned short* __restrict__ feat) // [P][CIN]
{
    __shared__ float sxyz[3 * SS];                       // 24 KB
    const int b = blockIdx.y;
    const int n = blockIdx.x * 256 + threadIdx.x;

    for (int i = threadIdx.x; i < 3 * SS; i += 256)
        sxyz[i] = xyz2[(size_t)b * 3 * SS + i];
    __syncthreads();

    const float px = xyz1[(size_t)b * 3 * NN + 0 * NN + n];
    const float py = xyz1[(size_t)b * 3 * NN + 1 * NN + n];
    const float pz = xyz1[(size_t)b * 3 * NN + 2 * NN + n];

    float d0 = 3.4e38f, dB = 3.4e38f, dC = 3.4e38f;
    int   i0 = 0, i1 = 0, i2 = 0;
    for (int s = 0; s < SS; ++s) {
        float dx = px - sxyz[s];
        float dy = py - sxyz[SS + s];
        float dz = pz - sxyz[2 * SS + s];
        float d = dx * dx + dy * dy + dz * dz;
        if (d < dC) {
            if (d < dB) {
                if (d < d0) { dC = dB; i2 = i1; dB = d0; i1 = i0; d0 = d; i0 = s; }
                else        { dC = dB; i2 = i1; dB = d;  i1 = s; }
            } else          { dC = d;  i2 = s; }
        }
    }
    float w0 = 1.0f / (d0 + DIST_EPS);
    float w1 = 1.0f / (dB + DIST_EPS);
    float w2 = 1.0f / (dC + DIST_EPS);
    float inv = 1.0f / (w0 + w1 + w2);
    w0 *= inv; w1 *= inv; w2 *= inv;

    const size_t prow = ((size_t)b * NN + n) * CIN;
    for (int c = 0; c < D1; ++c)
        feat[prow + c] = f2bf(points1[((size_t)b * D1 + c) * NN + n]);
    const float* p2b = points2 + (size_t)b * D2 * SS;
    for (int c = 0; c < D2; ++c) {
        const float* row = p2b + (size_t)c * SS;
        float v = w0 * row[i0] + w1 * row[i1] + w2 * row[i2];
        feat[prow + D1 + c] = f2bf(v);
    }
}

// ---------------- TDM descriptor helper ----------------
// 2-D tile load: tile_dim0=32 elems (bf16) per row, tile_dim1=128 rows,
// tensor row stride = K elems. LDS padding: 16B pad every 64B of data
// -> 80-byte LDS row stride (matches LDSK=40 ushorts).
#if USE_TDM
__device__ __forceinline__ void tdm_load_tile(const unsigned short* gptr,
                                              unsigned lds_off, int K) {
    unsigned long long ga = (unsigned long long)(size_t)gptr;
    u32x4 g0;
    g0.x = 1u;                                            // count=1 (valid D#)
    g0.y = lds_off;                                       // LDS byte address
    g0.z = (unsigned)(ga & 0xFFFFFFFFu);                  // global_addr[31:0]
    g0.w = (unsigned)((ga >> 32) & 0x01FFFFFFu)           // global_addr[56:32]
         | (2u << 30);                                    // type=2 ("image")
    const unsigned td1 = 1u << 20;                        // large dim1 (no OOB)
    i32x8 g1;
    g1[0] = (1 << 16)        // data_size = 2 bytes
          | (1 << 20)        // pad_enable
          | (3 << 22)        // pad_interval: 16 DWORDs (=64B)
          | (3 << 25);       // pad_amount:   4 DWORDs (=16B)
    g1[1] = (int)((K & 0xFFFF) << 16);                    // tensor_dim0 lo
    g1[2] = (int)(((unsigned)K >> 16) | ((td1 & 0xFFFFu) << 16));
    g1[3] = (int)((td1 >> 16) | (32u << 16));             // dim1 hi | tile_dim0=32
    g1[4] = 128;                                          // tile_dim1=128, tile_dim2=0
    g1[5] = K;                                            // tensor_dim0_stride lo
    g1[6] = 0;                                            // stride hi | dim1_stride lo
    g1[7] = 0;
    i32x4 z4 = {0, 0, 0, 0};
#if __clang_major__ >= 23
    i32x8 z8 = {0, 0, 0, 0, 0, 0, 0, 0};
    __builtin_amdgcn_tensor_load_to_lds(g0, g1, z4, z4, z8, 0);
#else
    __builtin_amdgcn_tensor_load_to_lds(g0, g1, z4, z4, 0);
#endif
}
#endif

// ---------------- bf16 WMMA GEMM ----------------
// C[M x 512] = A[M x K] (bf16) * W[512 x K]^T (bf16) + bias
// grid (512/128, M/128), block 256 (8 waves). Wave tile 32x64 (2x4 WMMAs).
#define LDSK 40   // 32 + 8 pad (ushorts) -> 80B row stride
__global__ __launch_bounds__(256)
void gemm_bf16_wmma(const unsigned short* __restrict__ A,
                    const unsigned short* __restrict__ W,   // [512][K]
                    const float* __restrict__ bias,
                    float* __restrict__ C, int K)
{
#if USE_TDM
    __shared__ unsigned short As[2][128 * LDSK];
    __shared__ unsigned short Bs[2][128 * LDSK];
#else
    __shared__ unsigned short As[1][128 * LDSK];
    __shared__ unsigned short Bs[1][128 * LDSK];
#endif

    const int tid   = threadIdx.x;
    const int wid   = tid >> 5;
    const int lane  = tid & 31;
    const int l16   = lane & 15;
    const int lhalf = lane >> 4;
    const int wm    = wid >> 1;     // 0..3 -> 32-row slice
    const int wn    = wid & 1;      // 0..1 -> 64-col slice
    const int pbase = blockIdx.y * 128;
    const int cbase = blockIdx.x * 128;

    v8f acc[2][4];
#pragma unroll
    for (int mt = 0; mt < 2; ++mt)
#pragma unroll
        for (int nt = 0; nt < 4; ++nt)
            acc[mt][nt] = (v8f)(0.0f);

    const int nsteps = K / 32;

#if USE_TDM
    // LDS byte addresses (generic shared ptr low 32 bits = LDS aperture offset)
    const unsigned asoff0 = (unsigned)(size_t)&As[0][0];
    const unsigned asoff1 = (unsigned)(size_t)&As[1][0];
    const unsigned bsoff0 = (unsigned)(size_t)&Bs[0][0];
    const unsigned bsoff1 = (unsigned)(size_t)&Bs[1][0];
    if (wid == 0) {   // one wave drives the tensor data mover
        tdm_load_tile(A + (size_t)pbase * K, asoff0, K);
        tdm_load_tile(W + (size_t)cbase * K, bsoff0, K);
    }
#endif

    for (int s = 0; s < nsteps; ++s) {
        const int k0 = s * 32;
#if USE_TDM
        const int buf = s & 1;
        if (wid == 0) {
            if (s + 1 < nsteps) {   // issue next tile into other buffer
                tdm_load_tile(A + (size_t)pbase * K + k0 + 32,
                              buf ? asoff0 : asoff1, K);
                tdm_load_tile(W + (size_t)cbase * K + k0 + 32,
                              buf ? bsoff0 : bsoff1, K);
                __builtin_amdgcn_s_wait_tensorcnt(2);  // current tile complete
            } else {
                __builtin_amdgcn_s_wait_tensorcnt(0);
            }
        }
        __syncthreads();
#else
        const int buf = 0;
#pragma unroll
        for (int i = 0; i < 2; ++i) {
            int chunk = tid + i * 256;        // 512 chunks of 16B
            int row = chunk >> 2, q = chunk & 3;
            *(uint4*)(As[0] + row * LDSK + q * 8) =
                *(const uint4*)(A + (size_t)(pbase + row) * K + k0 + q * 8);
            *(uint4*)(Bs[0] + row * LDSK + q * 8) =
                *(const uint4*)(W + (size_t)(cbase + row) * K + k0 + q * 8);
        }
        if (k0 + 32 < K) {
            __builtin_prefetch(A + (size_t)(pbase + (tid >> 2)) * K + k0 + 32 + (tid & 3) * 8, 0, 1);
            __builtin_prefetch(W + (size_t)(cbase + (tid >> 2)) * K + k0 + 32 + (tid & 3) * 8, 0, 1);
        }
        __syncthreads();
#endif

        // A fragments: lane = row, two 16B K-chunks per ISA 16-bit A layout
        v16bf afrag[2];
#pragma unroll
        for (int mt = 0; mt < 2; ++mt) {
            int row = wm * 32 + mt * 16 + l16;
            v8bf lo = *(const v8bf*)(As[buf] + row * LDSK + lhalf * 8);
            v8bf hi = *(const v8bf*)(As[buf] + row * LDSK + 16 + lhalf * 8);
            v16bf a;
#pragma unroll
            for (int j = 0; j < 8; ++j) { a[j] = lo[j]; a[j + 8] = hi[j]; }
            afrag[mt] = a;
        }
        // B fragments: lane = col, two 16B K-chunks (same K order as A)
        v16bf bfrag[4];
#pragma unroll
        for (int nt = 0; nt < 4; ++nt) {
            int col = wn * 64 + nt * 16 + l16;
            v8bf lo = *(const v8bf*)(Bs[buf] + col * LDSK + lhalf * 8);
            v8bf hi = *(const v8bf*)(Bs[buf] + col * LDSK + 16 + lhalf * 8);
            v16bf bfr;
#pragma unroll
            for (int j = 0; j < 8; ++j) { bfr[j] = lo[j]; bfr[j + 8] = hi[j]; }
            bfrag[nt] = bfr;
        }
#pragma unroll
        for (int mt = 0; mt < 2; ++mt)
#pragma unroll
            for (int nt = 0; nt < 4; ++nt)
                acc[mt][nt] = __builtin_amdgcn_wmma_f32_16x16x32_bf16(
                    false, afrag[mt], false, bfrag[nt],
                    (short)0, acc[mt][nt], false, false);
        __syncthreads();
    }

    // epilogue: + bias, scatter per ISA C/D layout (n = lane%16, m = hi*8 + r)
#pragma unroll
    for (int nt = 0; nt < 4; ++nt) {
        int c = cbase + wn * 64 + nt * 16 + l16;
        float bv = bias[c];
#pragma unroll
        for (int mt = 0; mt < 2; ++mt) {
#pragma unroll
            for (int r = 0; r < 8; ++r) {
                int row = pbase + wm * 32 + mt * 16 + lhalf * 8 + r;
                C[(size_t)row * HH + c] = acc[mt][nt][r] + bv;
            }
        }
    }
}

// ---------------- batch-norm helpers ----------------
__global__ void zero_stats_kernel(float* s) {
    if (threadIdx.x < 2 * HH) s[threadIdx.x] = 0.0f;
}

__global__ __launch_bounds__(512)
void bn_stats_kernel(const float* __restrict__ x,
                     float* __restrict__ sums) {  // [0..511]=sum, [512..1023]=sumsq
    const int c = threadIdx.x;
    float s = 0.0f, ss = 0.0f;
    const size_t base = (size_t)blockIdx.x * 256;
    for (int i = 0; i < 256; ++i) {
        float v = x[(base + i) * HH + c];
        s += v; ss += v * v;
    }
    atomicAdd(&sums[c], s);
    atomicAdd(&sums[HH + c], ss);
}

__global__ void bn_finalize_kernel(const float* __restrict__ sums,
                                   const float* __restrict__ g,
                                   const float* __restrict__ be,
                                   float* __restrict__ scale,
                                   float* __restrict__ shift) {
    int c = threadIdx.x;
    if (c >= HH) return;
    const float inv_cnt = 1.0f / (float)PP;
    float m = sums[c] * inv_cnt;
    float v = sums[HH + c] * inv_cnt - m * m;
    float sc = g[c] * rsqrtf(v + BN_EPS);
    scale[c] = sc;
    shift[c] = be[c] - m * sc;
}

__global__ __launch_bounds__(256)
void bn_relu_bf16_kernel(const float* __restrict__ x,
                         const float* __restrict__ scale,
                         const float* __restrict__ shift,
                         unsigned short* __restrict__ y) {
    const size_t total = (size_t)PP * HH;
    for (size_t i = (size_t)blockIdx.x * blockDim.x + threadIdx.x; i < total;
         i += (size_t)gridDim.x * blockDim.x) {
        int c = (int)(i & (HH - 1));
        float v = fmaf(x[i], scale[c], shift[c]);
        y[i] = f2bf(fmaxf(v, 0.0f));
    }
}

__global__ __launch_bounds__(256)
void bn_relu_transpose_out(const float* __restrict__ x,
                           const float* __restrict__ scale,
                           const float* __restrict__ shift,
                           float* __restrict__ out) {
    __shared__ float tile[64][65];
    const int pbase = blockIdx.x * 64;
    const int cbase = blockIdx.y * 64;
    for (int i = threadIdx.x; i < 64 * 64; i += 256) {
        int pr = i >> 6, cc = i & 63;
        float v = x[(size_t)(pbase + pr) * HH + cbase + cc];
        v = fmaf(v, scale[cbase + cc], shift[cbase + cc]);
        tile[pr][cc] = fmaxf(v, 0.0f);
    }
    __syncthreads();
    for (int i = threadIdx.x; i < 64 * 64; i += 256) {
        int cr = i >> 6, pc = i & 63;
        int p = pbase + pc;
        int b = p >> 13, n = p & (NN - 1);
        out[((size_t)b * HH + cbase + cr) * NN + n] = tile[pc][cr];
    }
}

// ---------------- host-side orchestration ----------------
extern "C" void kernel_launch(void* const* d_in, const int* in_sizes, int n_in,
                              void* d_out, int out_size, void* d_ws, size_t ws_size,
                              hipStream_t stream) {
    const float* xyz1    = (const float*)d_in[0];
    const float* xyz2    = (const float*)d_in[1];
    const float* points1 = (const float*)d_in[2];
    const float* points2 = (const float*)d_in[3];
    const float* w0      = (const float*)d_in[4];
    const float* b0      = (const float*)d_in[5];
    const float* g0      = (const float*)d_in[6];
    const float* be0     = (const float*)d_in[7];
    const float* w1      = (const float*)d_in[8];
    const float* b1      = (const float*)d_in[9];
    const float* g1      = (const float*)d_in[10];
    const float* be1     = (const float*)d_in[11];
    float* out = (float*)d_out;

    char* ws = (char*)d_ws;
    size_t off = 0;
    unsigned short* w0bf = (unsigned short*)(ws + off); off += (size_t)HH * CIN * 2;
    unsigned short* w1bf = (unsigned short*)(ws + off); off += (size_t)HH * HH * 2;
    unsigned short* feat = (unsigned short*)(ws + off); off += (size_t)PP * CIN * 2;
    float*          x0   = (float*)(ws + off);          off += (size_t)PP * HH * 4;
    unsigned short* x1bf = (unsigned short*)(ws + off); off += (size_t)PP * HH * 2;
    float*          x2   = x0;  // x0 dead after bn_relu_bf16 -> reuse region
    float* stats  = (float*)(ws + off); off += 2 * HH * 4;   // sum | sumsq
    float* scale0 = (float*)(ws + off); off += HH * 4;
    float* shift0 = (float*)(ws + off); off += HH * 4;
    float* scale1 = (float*)(ws + off); off += HH * 4;
    float* shift1 = (float*)(ws + off); off += HH * 4;

    // 1) weights -> bf16
    cvt_bf16_kernel<<<(HH * CIN + 255) / 256, 256, 0, stream>>>(w0, w0bf, HH * CIN);
    cvt_bf16_kernel<<<(HH * HH + 255) / 256, 256, 0, stream>>>(w1, w1bf, HH * HH);

    // 2) 3-NN interpolation + concat -> feat bf16 [P][768]
    knn_interp_kernel<<<dim3(NN / 256, BB), 256, 0, stream>>>(
        xyz1, xyz2, points1, points2, feat);

    // 3) layer 0: GEMM + bias
    gemm_bf16_wmma<<<dim3(HH / 128, PP / 128), 256, 0, stream>>>(
        feat, w0bf, b0, x0, CIN);

    // 4) BN stats + apply + ReLU -> bf16
    zero_stats_kernel<<<1, 1024, 0, stream>>>(stats);
    bn_stats_kernel<<<PP / 256, 512, 0, stream>>>(x0, stats);
    bn_finalize_kernel<<<1, 512, 0, stream>>>(stats, g0, be0, scale0, shift0);
    bn_relu_bf16_kernel<<<2048, 256, 0, stream>>>(x0, scale0, shift0, x1bf);

    // 5) layer 1: GEMM + bias
    gemm_bf16_wmma<<<dim3(HH / 128, PP / 128), 256, 0, stream>>>(
        x1bf, w1bf, b1, x2, HH);

    // 6) BN stats + apply + ReLU + transpose to (B,H,N)
    zero_stats_kernel<<<1, 1024, 0, stream>>>(stats);
    bn_stats_kernel<<<PP / 256, 512, 0, stream>>>(x2, stats);
    bn_finalize_kernel<<<1, 512, 0, stream>>>(stats, g1, be1, scale1, shift1);
    bn_relu_transpose_out<<<dim3(PP / 64, HH / 64), 256, 0, stream>>>(
        x2, scale1, shift1, out);
}
